// TransformerBlock_De_87368224735424
// MI455X (gfx1250) — matrix operand order repaired and checked by reading, not compiled, not run on hardware
//
#include <hip/hip_runtime.h>
#include <hip/hip_bf16.h>
#include <stdint.h>

#define DIMN   512
#define HEADS  8
#define DKH    64
#define BATCH  4
#define SEQ    2048
#define MROWS  (BATCH*SEQ)   // 8192

typedef __attribute__((ext_vector_type(16))) __bf16 v16bf;
typedef __attribute__((ext_vector_type(8)))  float  v8f;

__device__ __forceinline__ unsigned short f32_bf16(float f) {
  unsigned int u = __float_as_uint(f);
  u += 0x7FFFu + ((u >> 16) & 1u);          // round-to-nearest-even
  return (unsigned short)(u >> 16);
}

__device__ __forceinline__ v16bf make_frag(uint4 lo, uint4 hi) {
  union { uint4 u[2]; v16bf v; } t;
  t.u[0] = lo; t.u[1] = hi;
  return t.v;
}

// CDNA5 async DMA: 16B/lane global -> LDS, tracked by ASYNCcnt.
// lds_off = LDS byte address (low 32 bits of generic shared pointer).
__device__ __forceinline__ void async_ld16(unsigned lds_off, const void* gaddr) {
  asm volatile("global_load_async_to_lds_b128 %0, %1, off"
               :: "v"(lds_off), "v"(gaddr) : "memory");
}
__device__ __forceinline__ void wait_async0() {
  asm volatile("s_wait_asynccnt 0x0" ::: "memory");
}

// ---------------------------------------------------------------------------
// Prep: pos gather + add, bf16 conversion of xp and vis_data
// ---------------------------------------------------------------------------
__global__ __launch_bounds__(256)
void prep_inputs(const float* __restrict__ x, const float* __restrict__ vis,
                 const int* __restrict__ c1, const int* __restrict__ c2,
                 const float* __restrict__ e1, const float* __restrict__ e2,
                 unsigned short* __restrict__ xp_bf,
                 unsigned short* __restrict__ vis_bf)
{
  size_t idx = (size_t)blockIdx.x * 256 + threadIdx.x;   // < 8192*512
  int row = (int)(idx >> 9);
  int c   = (int)(idx & 511);
  float pos = (c < 256) ? e1[(size_t)c1[row] * 256 + c]
                        : e2[(size_t)c2[row] * 256 + (c - 256)];
  xp_bf[idx]  = f32_bf16(x[idx] + pos);
  vis_bf[idx] = f32_bf16(vis[idx]);
}

// W[512][512] (row k, col n) -> WT[512][512] (row n, col k) in bf16
__global__ __launch_bounds__(256)
void weight_conv(const float* __restrict__ W, unsigned short* __restrict__ WT)
{
  int idx = blockIdx.x * 256 + threadIdx.x;   // 512*512
  int n = idx >> 9, k = idx & 511;
  WT[idx] = f32_bf16(W[(size_t)k * DIMN + n]);
}

// ---------------------------------------------------------------------------
// bf16 GEMM:  Y[M,N] = X[M,K] * WT[N,K]^T + bias   (M=8192, N=K=512)
// block tile 128x64, 8 waves, wave tile 16x64; K step 32.
// Double-buffered LDS fed by GLOBAL_LOAD_ASYNC_TO_LDS_B128 (ASYNCcnt).
// mode 0: store bf16 to [B,H,S,DK]   (Q/K; Q pre-scaled by 1/sqrt(DK))
// mode 1: store bf16 to [B,H,DK,S]   (V transposed)
// mode 2: f32 + leaky_relu(0.2) row-major
// ---------------------------------------------------------------------------
__global__ __launch_bounds__(256)
void gemm_bf16(const unsigned short* __restrict__ X,
               const unsigned short* __restrict__ WT,
               const float* __restrict__ bias,
               void* __restrict__ out, int mode, float scale)
{
  __shared__ alignas(16) unsigned short As[2][128 * 40];
  __shared__ alignas(16) unsigned short Bs[2][64 * 40];

  const int tid  = threadIdx.x;
  const int wave = tid >> 5, lane = tid & 31;
  const int m0 = blockIdx.x * 128, n0 = blockIdx.y * 64;
  const int lrow = lane & 15;
  const int hg   = lane >> 4;
  const int g    = hg * 8;

  // per-thread staging coordinates (16B chunks)
  const int ra0 = tid >> 2,         qa0 = tid & 3;          // A chunk 0
  const int ra1 = (tid + 256) >> 2, qa1 = tid & 3;          // A chunk 1
  const int rb  = tid >> 2,         qb  = tid & 3;          // B chunk

  auto stage = [&](int buf, int k0) {
    async_ld16((unsigned)(size_t)&As[buf][ra0 * 40 + qa0 * 8],
               X + (size_t)(m0 + ra0) * DIMN + k0 + qa0 * 8);
    async_ld16((unsigned)(size_t)&As[buf][ra1 * 40 + qa1 * 8],
               X + (size_t)(m0 + ra1) * DIMN + k0 + qa1 * 8);
    async_ld16((unsigned)(size_t)&Bs[buf][rb * 40 + qb * 8],
               WT + (size_t)(n0 + rb) * DIMN + k0 + qb * 8);
  };

  v8f acc[4] = {};

  stage(0, 0);
  int buf = 0;
  for (int k0 = 0; k0 < DIMN; k0 += 32) {
    wait_async0();       // my async writes (to buf) have landed in LDS
    __syncthreads();     // everyone's writes landed; prior reads of buf^1 done
    if (k0 + 32 < DIMN) stage(buf ^ 1, k0 + 32);   // overlap DMA with WMMAs

    // issue ALL fragment loads first so ds_loads batch with partial waits
    const int arow = wave * 16 + lrow;
    const int kk = hg * 16;
    v16bf afrag = make_frag(*(const uint4*)(&As[buf][arow * 40 + kk]),
                            *(const uint4*)(&As[buf][arow * 40 + kk + 8]));
    v16bf bfr[4];
#pragma unroll
    for (int nt = 0; nt < 4; ++nt) {
      const int brow = nt * 16 + lrow;
      bfr[nt] = make_frag(*(const uint4*)(&Bs[buf][brow * 40 + kk]),
                          *(const uint4*)(&Bs[buf][brow * 40 + kk + 8]));
    }
#pragma unroll
    for (int nt = 0; nt < 4; ++nt)
      acc[nt] = __builtin_amdgcn_wmma_f32_16x16x32_bf16(
          false, afrag, false, bfr[nt], (short)0, acc[nt], false, false);
    buf ^= 1;
  }

  // epilogue
#pragma unroll
  for (int nt = 0; nt < 4; ++nt) {
    const int n = n0 + nt * 16 + lrow;
    const float bv = bias[n];
#pragma unroll
    for (int r = 0; r < 8; ++r) {
      const int m = m0 + wave * 16 + r + hg * 8;
      float v = (acc[nt][r] + bv) * scale;
      if (mode == 0) {                       // [B,H,S,DK] bf16
        int b = m >> 11, s = m & 2047;
        int h = n >> 6,  d = n & 63;
        ((unsigned short*)out)[((((size_t)b * HEADS + h) * SEQ + s) << 6) + d] = f32_bf16(v);
      } else if (mode == 1) {                // [B,H,DK,S] bf16 (V^T)
        int b = m >> 11, s = m & 2047;
        int h = n >> 6,  d = n & 63;
        ((unsigned short*)out)[((((size_t)b * HEADS + h) * DKH + d) << 11) + s] = f32_bf16(v);
      } else {                               // f32 + leaky relu
        float y = v > 0.f ? v : 0.2f * v;
        ((float*)out)[(size_t)m * DIMN + n] = y;
      }
    }
  }
}

// ---------------------------------------------------------------------------
// Flash attention: one wave per 16-query tile per (b,h). Streaming softmax.
// Q,K: [B,H,S,DK] bf16 (Q pre-scaled 1/8). V: [B,H,DK,S] bf16. Out: bf16 [M,DIM]
// ---------------------------------------------------------------------------
__global__ __launch_bounds__(256)
void flash_attn(const unsigned short* __restrict__ Q,
                const unsigned short* __restrict__ K,
                const unsigned short* __restrict__ VT,
                unsigned short* __restrict__ O)
{
  __shared__ alignas(16) unsigned short P[8][16 * 40];  // per-wave P tile

  const int tid  = threadIdx.x;
  const int wave = tid >> 5, lane = tid & 31;
  const int tile = blockIdx.x * 8 + wave;    // 4096 query tiles total
  const int qt = tile & 127;                 // S/16 = 128
  const int bh = tile >> 7;                  // 0..31
  const size_t base_qk = (size_t)bh * SEQ * DKH;
  const size_t base_vt = (size_t)bh * DKH * SEQ;
  const int lrow = lane & 15;
  const int hg   = lane >> 4;
  const int g    = hg * 8;

  // Q A-fragments for both 32-wide K-chunks of DK=64
  v16bf qf[2];
  {
    const unsigned short* qrow = Q + base_qk + (size_t)(qt * 16 + lrow) * DKH;
#pragma unroll
    for (int kc = 0; kc < 2; ++kc)
      qf[kc] = make_frag(*(const uint4*)(qrow + kc * 32 + hg * 16),
                         *(const uint4*)(qrow + kc * 32 + hg * 16 + 8));
  }

  float mrow[8], lsum[8];
#pragma unroll
  for (int r = 0; r < 8; ++r) { mrow[r] = -3.0e38f; lsum[r] = 0.f; }
  v8f o[4] = {};

  unsigned short* pw = &P[wave][0];

  for (int j0 = 0; j0 < SEQ; j0 += 32) {
    // load all 4 K fragments up front (batched global_load_b128s)
    v16bf kf[4];
#pragma unroll
    for (int jt = 0; jt < 2; ++jt) {
      const unsigned short* krow =
          K + base_qk + (size_t)(j0 + jt * 16 + lrow) * DKH;
#pragma unroll
      for (int kc = 0; kc < 2; ++kc) {
        const int kk = kc * 32 + hg * 16;
        kf[jt * 2 + kc] = make_frag(*(const uint4*)(krow + kk),
                                    *(const uint4*)(krow + kk + 8));
      }
    }
    v8f s0 = {}, s1 = {};
    s0 = __builtin_amdgcn_wmma_f32_16x16x32_bf16(false, qf[0], false, kf[0], (short)0, s0, false, false);
    s0 = __builtin_amdgcn_wmma_f32_16x16x32_bf16(false, qf[1], false, kf[1], (short)0, s0, false, false);
    s1 = __builtin_amdgcn_wmma_f32_16x16x32_bf16(false, qf[0], false, kf[2], (short)0, s1, false, false);
    s1 = __builtin_amdgcn_wmma_f32_16x16x32_bf16(false, qf[1], false, kf[3], (short)0, s1, false, false);

    // streaming softmax update (row = r + 8*hg; cols across 16-lane group)
#pragma unroll
    for (int r = 0; r < 8; ++r) {
      float mx = fmaxf(s0[r], s1[r]);
      mx = fmaxf(mx, __shfl_xor(mx, 1, 32));
      mx = fmaxf(mx, __shfl_xor(mx, 2, 32));
      mx = fmaxf(mx, __shfl_xor(mx, 4, 32));
      mx = fmaxf(mx, __shfl_xor(mx, 8, 32));
      float mnew = fmaxf(mrow[r], mx);
      float corr = __expf(mrow[r] - mnew);
      mrow[r] = mnew;
      float p0 = __expf(s0[r] - mnew);
      float p1 = __expf(s1[r] - mnew);
      float rs = p0 + p1;
      rs += __shfl_xor(rs, 1, 32);
      rs += __shfl_xor(rs, 2, 32);
      rs += __shfl_xor(rs, 4, 32);
      rs += __shfl_xor(rs, 8, 32);
      lsum[r] = lsum[r] * corr + rs;
      o[0][r] *= corr; o[1][r] *= corr; o[2][r] *= corr; o[3][r] *= corr;
      s0[r] = p0; s1[r] = p1;
    }

    // C-layout P -> LDS (bf16), then reload as A-fragment
#pragma unroll
    for (int r = 0; r < 8; ++r) {
      const int pr = r + hg * 8;
      pw[pr * 40 + lrow]      = f32_bf16(s0[r]);
      pw[pr * 40 + 16 + lrow] = f32_bf16(s1[r]);
    }
    asm volatile("s_wait_dscnt 0" ::: "memory");  // same-wave cross-lane LDS
    v16bf pf = make_frag(*(const uint4*)(&pw[lrow * 40 + hg * 16]),
                         *(const uint4*)(&pw[lrow * 40 + hg * 16 + 8]));

    // load all 4 V fragments, then chain the 4 PV WMMAs
    v16bf vf[4];
#pragma unroll
    for (int dt = 0; dt < 4; ++dt) {
      const unsigned short* vrow =
          VT + base_vt + (size_t)(dt * 16 + lrow) * SEQ + j0 + hg * 16;
      vf[dt] = make_frag(*(const uint4*)(vrow),
                         *(const uint4*)(vrow + 8));
    }
#pragma unroll
    for (int dt = 0; dt < 4; ++dt)
      o[dt] = __builtin_amdgcn_wmma_f32_16x16x32_bf16(
          false, pf, false, vf[dt], (short)0, o[dt], false, false);
  }

  // epilogue: O /= l, store bf16 [B,S,DIM]
  const int b = bh >> 3, h = bh & 7;
#pragma unroll
  for (int dt = 0; dt < 4; ++dt) {
#pragma unroll
    for (int r = 0; r < 8; ++r) {
      const int srow = qt * 16 + r + hg * 8;
      float val = o[dt][r] / lsum[r];
      O[(size_t)(b * SEQ + srow) * DIMN + h * DKH + dt * 16 + lrow] = f32_bf16(val);
    }
  }
}

// ---------------------------------------------------------------------------
// LayerNorm over rows of 512; block(128) per row; to_f32=0 -> bf16 out
// ---------------------------------------------------------------------------
__global__ __launch_bounds__(128)
void layernorm_k(const float* __restrict__ X, const float* __restrict__ gamma,
                 const float* __restrict__ beta, void* __restrict__ out,
                 int to_f32)
{
  __shared__ float red[8];
  const int tid = threadIdx.x;
  const int row = blockIdx.x;
  float4 v = ((const float4*)(X + (size_t)row * DIMN))[tid];
  float s = v.x + v.y + v.z + v.w;
  float q = v.x * v.x + v.y * v.y + v.z * v.z + v.w * v.w;
#pragma unroll
  for (int m = 1; m < 32; m <<= 1) {
    s += __shfl_xor(s, m, 32);
    q += __shfl_xor(q, m, 32);
  }
  if ((tid & 31) == 0) { red[tid >> 5] = s; red[4 + (tid >> 5)] = q; }
  __syncthreads();
  s = red[0] + red[1] + red[2] + red[3];
  q = red[4] + red[5] + red[6] + red[7];
  const float mean = s * (1.f / DIMN);
  const float var  = q * (1.f / DIMN) - mean * mean;
  const float rstd = rsqrtf(var + 1e-5f);
  const int c = tid * 4;
  float ys[4] = { v.x, v.y, v.z, v.w };
  if (to_f32) {
    float* o = (float*)out + (size_t)row * DIMN + c;
#pragma unroll
    for (int i = 0; i < 4; ++i)
      o[i] = (ys[i] - mean) * rstd * gamma[c + i] + beta[c + i];
  } else {
    unsigned short* o = (unsigned short*)out + (size_t)row * DIMN + c;
#pragma unroll
    for (int i = 0; i < 4; ++i)
      o[i] = f32_bf16((ys[i] - mean) * rstd * gamma[c + i] + beta[c + i]);
  }
}

// ---------------------------------------------------------------------------
extern "C" void kernel_launch(void* const* d_in, const int* in_sizes, int n_in,
                              void* d_out, int out_size, void* d_ws, size_t ws_size,
                              hipStream_t stream)
{
  (void)in_sizes; (void)n_in; (void)out_size; (void)ws_size;
  const float* x   = (const float*)d_in[0];
  const float* vis = (const float*)d_in[1];
  const int*   c1  = (const int*)d_in[2];
  const int*   c2  = (const int*)d_in[3];
  const float* e1  = (const float*)d_in[4];
  const float* e2  = (const float*)d_in[5];
  const float* wq  = (const float*)d_in[6];
  const float* bq  = (const float*)d_in[7];
  const float* wk  = (const float*)d_in[8];
  const float* bk  = (const float*)d_in[9];
  const float* wv  = (const float*)d_in[10];
  const float* bv  = (const float*)d_in[11];
  const float* w1  = (const float*)d_in[12];
  const float* b1  = (const float*)d_in[13];
  const float* g1  = (const float*)d_in[14];
  const float* be1 = (const float*)d_in[15];
  const float* w2  = (const float*)d_in[16];
  const float* b2  = (const float*)d_in[17];
  const float* g2  = (const float*)d_in[18];
  const float* be2 = (const float*)d_in[19];

  char* ws = (char*)d_ws;
  const size_t MB = 1ull << 20;
  unsigned short* vis_bf = (unsigned short*)(ws + 0);        //  8 MB
  unsigned short* xp_bf  = (unsigned short*)(ws + 8  * MB);  //  8 MB
  unsigned short* wt     = (unsigned short*)(ws + 16 * MB);  //  2.5 MB
  unsigned short* q_bf   = (unsigned short*)(ws + 19 * MB);  //  8 MB
  unsigned short* k_bf   = (unsigned short*)(ws + 27 * MB);  //  8 MB
  unsigned short* vt_bf  = (unsigned short*)(ws + 35 * MB);  //  8 MB
  unsigned short* a_bf   = (unsigned short*)(ws + 43 * MB);  //  8 MB
  float*          h1     = (float*)(ws + 19 * MB);           // 16 MB (reuse q,k)
  unsigned short* h1n    = (unsigned short*)(ws + 35 * MB);  //  8 MB (reuse vt)
  float*          h2     = (float*)(ws + 0);                 // 16 MB (reuse vis,xp)

  const int WSZ = DIMN * DIMN;

  prep_inputs<<<(MROWS * DIMN) / 256, 256, 0, stream>>>(x, vis, c1, c2, e1, e2,
                                                        xp_bf, vis_bf);
  weight_conv<<<WSZ / 256, 256, 0, stream>>>(wq, wt + 0 * WSZ);
  weight_conv<<<WSZ / 256, 256, 0, stream>>>(wk, wt + 1 * WSZ);
  weight_conv<<<WSZ / 256, 256, 0, stream>>>(wv, wt + 2 * WSZ);
  weight_conv<<<WSZ / 256, 256, 0, stream>>>(w1, wt + 3 * WSZ);
  weight_conv<<<WSZ / 256, 256, 0, stream>>>(w2, wt + 4 * WSZ);

  dim3 ggrid(MROWS / 128, DIMN / 64);
  gemm_bf16<<<ggrid, 256, 0, stream>>>(vis_bf, wt + 0 * WSZ, bq, q_bf, 0, 0.125f);
  gemm_bf16<<<ggrid, 256, 0, stream>>>(vis_bf, wt + 1 * WSZ, bk, k_bf, 0, 1.0f);
  gemm_bf16<<<ggrid, 256, 0, stream>>>(xp_bf,  wt + 2 * WSZ, bv, vt_bf, 1, 1.0f);

  flash_attn<<<(BATCH * HEADS * (SEQ / 16)) / 8, 256, 0, stream>>>(q_bf, k_bf,
                                                                   vt_bf, a_bf);

  gemm_bf16<<<ggrid, 256, 0, stream>>>(a_bf, wt + 3 * WSZ, b1, h1, 2, 1.0f);
  layernorm_k<<<MROWS, 128, 0, stream>>>(h1, g1, be1, h1n, 0);
  gemm_bf16<<<ggrid, 256, 0, stream>>>(h1n, wt + 4 * WSZ, b2, h2, 2, 1.0f);
  layernorm_k<<<MROWS, 128, 0, stream>>>(h2, g2, be2, d_out, 1);
}
